// Conv1DNet_35167192220279
// MI455X (gfx1250) — compile-verified, matrix operand order checked by reading, not run on hardware
//
#include <hip/hip_runtime.h>
#include <hip/hip_bf16.h>

// ---------------------------------------------------------------------------
// MI455X (gfx1250) implementation.
//   - x reduction: HBM-bound (524 MB @ 23.3 TB/s ~ 22.5 us floor) -> wide
//     float4 streaming reduction, 4096 blocks.
//   - conv1 (64x64x3 over L=1024, B=128) and the final 128x1024x512 GEMM use
//     v_wmma_f32_16x16x32_f16 (f16 in, f32 accumulate). All intermediates are
//     L2-resident (~67 MB << 192 MB L2).
// ---------------------------------------------------------------------------

typedef __attribute__((ext_vector_type(16))) _Float16 v16h;
typedef __attribute__((ext_vector_type(8)))  _Float16 v8h;
typedef __attribute__((ext_vector_type(4)))  _Float16 v4h;
typedef __attribute__((ext_vector_type(8)))  float    v8f;
typedef __attribute__((ext_vector_type(4)))  float    v4f;

#define B_SZ     128
#define PROJ_N   1024
#define CH       64
#define LPAD     1026          // 1024 + 1 halo row each side
#define IN_DIM_V4   255750     // 1023000 / 4
#define FULL_V4     255000     // 1020000 / 4

// ---- workspace layout (bytes) ----
static constexpr size_t OFF_PART = 0;                                   //  4096 f32
static constexpr size_t OFF_W1   = 16384;                               // 12288 f16 (w1 as [k][co][ci])
static constexpr size_t OFF_WOUT = 40960;                               // 512*1024 f16 (proj_out transposed [n][k])
static constexpr size_t OFF_A0T  = 1089536;                            // 128*1026*64 f16
static constexpr size_t OFF_C1   = 17899520;                           // 128*1024*64 f32 (raw conv1, pre-GN)
static constexpr size_t OFF_A1T  = 51453952;                           // 128*1026*64 f16
static constexpr size_t OFF_H    = 68263936;                           // 128*1024 f16
// total ~ 68.5 MB

__device__ __forceinline__ float mishf(float x) {
    float sp = (x > 20.f) ? x : log1pf(__expf(x));
    return x * tanhf(sp);
}

// Per-lane f16 WMMA fragment: two contiguous 8-half runs (K = +0..7, +16..23)
// relative to `p` (16-byte aligned) -> two global_load_b128.
__device__ __forceinline__ v16h load_frag(const _Float16* __restrict__ p) {
    v8h lo = *(const v8h*)(p);
    v8h hi = *(const v8h*)(p + 16);
    v16h r;
#pragma unroll
    for (int i = 0; i < 8; ++i) { r[i] = lo[i]; r[i + 8] = hi[i]; }
    return r;
}

// ---------------------------------------------------------------------------
// Weight prep: conv1_w (64,64,3) f32 -> w1f16 [k][co][ci] f16
// ---------------------------------------------------------------------------
__global__ void prep_w1_kernel(const float* __restrict__ w1, _Float16* __restrict__ w1f16) {
    int idx = blockIdx.x * 256 + threadIdx.x;          // 3*64*64 = 12288
    if (idx < 3 * CH * CH) {
        int k = idx >> 12;
        int rem = idx & 4095;
        int co = rem >> 6;
        int ci = rem & 63;
        w1f16[idx] = (_Float16)w1[(co * CH + ci) * 3 + k];
    }
}

// proj_out_weight (1024,512) f32 -> woutT [n][k] f16
__global__ void prep_wout_kernel(const float* __restrict__ wout, _Float16* __restrict__ woutT) {
    int idx = blockIdx.x * 256 + threadIdx.x;          // 512*1024
    int n = idx >> 10;
    int k = idx & 1023;
    woutT[idx] = (_Float16)wout[(size_t)k * 512 + n];
}

// ---------------------------------------------------------------------------
// Streaming reduction of x: partials[r*32+c] = weighted partial sum
// ---------------------------------------------------------------------------
__global__ void reduce_x_kernel(const float* __restrict__ x, float* __restrict__ partials) {
    const int r = blockIdx.x >> 5;   // row 0..127
    const int c = blockIdx.x & 31;   // slice 0..31
    const int t = threadIdx.x;
    const float4* xr = (const float4*)(x + (size_t)r * 1023000);
    float sA = 0.f, sB = 0.f;
    for (int i = c * 256 + t; i < IN_DIM_V4; i += 32 * 256) {
        float4 v = xr[i];
        float s = (v.x + v.y) + (v.z + v.w);
        if (i < FULL_V4) sA += s; else sB += s;
    }
    __shared__ float red[256];
    red[t] = sA * (1.f / 10000.f) + sB * (1.f / 3000.f);
    __syncthreads();
    for (int off = 128; off > 0; off >>= 1) {
        if (t < off) red[t] += red[t + off];
        __syncthreads();
    }
    if (t == 0) partials[blockIdx.x] = red[0];
}

// ---------------------------------------------------------------------------
// Stage 0: s -> h0 -> conv0 -> gn0 -> mish -> a0t (transposed f16, padded)
// one block per batch element
// ---------------------------------------------------------------------------
__global__ void stage0_kernel(const float* __restrict__ partials,
                              const float* __restrict__ piw, const float* __restrict__ pib,
                              const float* __restrict__ w0,  const float* __restrict__ b0,
                              const float* __restrict__ g0w, const float* __restrict__ g0b,
                              _Float16* __restrict__ a0t) {
    const int b = blockIdx.x;
    const int t = threadIdx.x;
    __shared__ float h0[LPAD];
    __shared__ float gsum[8], gsq[8], gmean[8], ginv[8];
    __shared__ float s_sh;
    if (t == 0) {
        float a = 0.f;
        for (int c = 0; c < 32; ++c) a += partials[b * 32 + c];
        s_sh = a;
        h0[0] = 0.f; h0[LPAD - 1] = 0.f;
    }
    if (t < 8) { gsum[t] = 0.f; gsq[t] = 0.f; }
    __syncthreads();
    const float s = s_sh;
    for (int l = t; l < PROJ_N; l += 256) h0[l + 1] = s * piw[l] + pib[l];
    __syncthreads();

    const int co = t >> 2;              // channel 0..63
    const int lb = (t & 3) * 256;       // quarter of the length
    const float w0a = w0[co * 3 + 0], w0b = w0[co * 3 + 1], w0c = w0[co * 3 + 2];
    const float bb = b0[co];
    float lsum = 0.f, lsq = 0.f;
    for (int i = 0; i < 256; ++i) {
        int l = lb + i;
        float v = w0a * h0[l] + w0b * h0[l + 1] + w0c * h0[l + 2] + bb;
        lsum += v; lsq += v * v;
    }
    atomicAdd(&gsum[co >> 3], lsum);
    atomicAdd(&gsq[co >> 3], lsq);
    __syncthreads();
    if (t < 8) {
        float m = gsum[t] * (1.f / 8192.f);
        float v = gsq[t] * (1.f / 8192.f) - m * m;
        gmean[t] = m; ginv[t] = rsqrtf(v + 1e-5f);
    }
    __syncthreads();
    const float gm = gmean[co >> 3], gi = ginv[co >> 3];
    const float gw = g0w[co], gb = g0b[co];
    for (int i = 0; i < 256; ++i) {
        int l = lb + i;
        float v = w0a * h0[l] + w0b * h0[l + 1] + w0c * h0[l + 2] + bb;
        v = mishf((v - gm) * gi * gw + gb);
        a0t[((size_t)b * LPAD + (l + 1)) * CH + co] = (_Float16)v;
    }
    if (t < CH) {   // zero halo rows
        a0t[((size_t)b * LPAD) * CH + t]              = (_Float16)0.f;
        a0t[((size_t)b * LPAD + LPAD - 1) * CH + t]   = (_Float16)0.f;
    }
}

// ---------------------------------------------------------------------------
// conv1 as WMMA GEMM (C = sum_k W1k[64x64] * shift_k(A0)[64x1024]) + bias,
// fused GroupNorm(8 groups) + mish, output transposed f16 a1t (padded).
// One block per batch element; 8 waves: wave = (M-tile 0..3) x (N-half 0..1).
// ---------------------------------------------------------------------------
__global__ void conv1_wmma_kernel(const _Float16* __restrict__ a0t,
                                  const _Float16* __restrict__ w1f16,
                                  const float* __restrict__ c1b,
                                  const float* __restrict__ g1w,
                                  const float* __restrict__ g1b,
                                  float*   __restrict__ c1raw,
                                  _Float16* __restrict__ a1t) {
    const int b    = blockIdx.x;
    const int t    = threadIdx.x;
    const int lane = t & 31;
    const int w    = t >> 5;

    __shared__ float gsum[8], gsq[8], gmean[8], ginv[8];
    if (t < 8) { gsum[t] = 0.f; gsq[t] = 0.f; }
    __syncthreads();

    const int mt    = w & 3;               // M-tile: out channels 16*mt..
    const int nhalf = w >> 2;              // N tiles 0..31 or 32..63
    const int m0    = mt * 16;
    const int h0    = (lane >> 4) * 8;     // per-lane K base pattern
    const int hi8   = h0;                  // C/D row offset for lanes 16..31
    const int rowA  = m0 + (lane & 15);    // A-matrix row (co)
    const int nB    = lane & 15;           // B-matrix column (l within tile)

    // 6 A fragments (3 taps x 2 k-steps), loaded once per wave.
    v16h afr[6];
#pragma unroll
    for (int k = 0; k < 3; ++k)
#pragma unroll
        for (int ks = 0; ks < 2; ++ks)
            afr[k * 2 + ks] = load_frag(w1f16 + ((size_t)(k * CH + rowA)) * CH + ks * 32 + h0);

    float bias[8];
#pragma unroll
    for (int j = 0; j < 8; ++j) bias[j] = c1b[m0 + hi8 + j];

    float lsum = 0.f, lsq = 0.f;
    for (int nt = 0; nt < 32; ++nt) {
        const int l0 = (nhalf * 32 + nt) * 16;
        v8f acc = {};
#pragma unroll
        for (int k = 0; k < 3; ++k) {
            const size_t rbase = ((size_t)b * LPAD + (l0 + nB + k)) * CH;   // padded row index
#pragma unroll
            for (int ks = 0; ks < 2; ++ks) {
                v16h bf = load_frag(a0t + rbase + ks * 32 + h0);
                acc = __builtin_amdgcn_wmma_f32_16x16x32_f16(
                        false, afr[k * 2 + ks], false, bf, (short)0, acc, false, false);
            }
        }
        const size_t obase = ((size_t)b * PROJ_N + (l0 + nB)) * CH + m0 + hi8;
#pragma unroll
        for (int j = 0; j < 8; ++j) {
            float v = acc[j] + bias[j];
            c1raw[obase + j] = v;
            lsum += v; lsq += v * v;
        }
    }
    // all 8 channels this lane touched belong to one GN group:
    const int g = 2 * mt + (lane >> 4);
    atomicAdd(&gsum[g], lsum);
    atomicAdd(&gsq[g], lsq);
    __threadfence_block();
    __syncthreads();
    if (t < 8) {
        float m = gsum[t] * (1.f / 8192.f);
        float v = gsq[t] * (1.f / 8192.f) - m * m;
        gmean[t] = m; ginv[t] = rsqrtf(v + 1e-5f);
    }
    __syncthreads();

    // normalize + mish -> a1t [l+1][co] f16 (vectorized over 4 channels)
    for (int i = 0; i < 64; ++i) {
        int flat = (t + i * 256) * 4;        // 0 .. 65532
        int l  = flat >> 6;
        int co = flat & 63;
        int g2 = co >> 3;
        v4f vv = *(const v4f*)(c1raw + ((size_t)b * PROJ_N + l) * CH + co);
        const float gm = gmean[g2], gi = ginv[g2];
        v4h hv;
#pragma unroll
        for (int j = 0; j < 4; ++j) {
            float v = ((float)vv[j] - gm) * gi * g1w[co + j] + g1b[co + j];
            hv[j] = (_Float16)mishf(v);
        }
        *(v4h*)(a1t + ((size_t)b * LPAD + (l + 1)) * CH + co) = hv;
    }
    if (t < CH) {   // zero halo rows
        a1t[((size_t)b * LPAD) * CH + t]            = (_Float16)0.f;
        a1t[((size_t)b * LPAD + LPAD - 1) * CH + t] = (_Float16)0.f;
    }
}

// ---------------------------------------------------------------------------
// conv2 (64 -> 1 channel, 3 taps): h[b][l] -> f16 for final WMMA GEMM
// ---------------------------------------------------------------------------
__global__ void conv2_kernel(const _Float16* __restrict__ a1t,
                             const float* __restrict__ w2, const float* __restrict__ b2,
                             _Float16* __restrict__ hbuf) {
    const int idx = blockIdx.x * 256 + threadIdx.x;   // 128*1024
    const int b = idx >> 10, l = idx & 1023;
    float acc = b2[0];
#pragma unroll
    for (int k = 0; k < 3; ++k) {
        const _Float16* row = a1t + ((size_t)b * LPAD + (l + k)) * CH;
#pragma unroll 8
        for (int ci = 0; ci < CH; ++ci)
            acc += (float)row[ci] * w2[ci * 3 + k];
    }
    hbuf[idx] = (_Float16)acc;
}

// ---------------------------------------------------------------------------
// Output GEMM: out[128,512] = H[128,1024] @ Wout[1024,512] + bias  (WMMA)
// grid = 32 N-tiles; 8 waves/block = 8 M-tiles; 32 K-steps of 32.
// ---------------------------------------------------------------------------
__global__ void out_gemm_kernel(const _Float16* __restrict__ hbuf,
                                const _Float16* __restrict__ woutT,
                                const float* __restrict__ obias,
                                float* __restrict__ out) {
    const int nt   = blockIdx.x;        // 0..31
    const int t    = threadIdx.x;
    const int lane = t & 31;
    const int mt   = t >> 5;            // 0..7
    const int m0 = mt * 16, n0 = nt * 16;
    const int h0  = (lane >> 4) * 8;
    const int rA  = m0 + (lane & 15);   // batch row for A
    const int cB  = n0 + (lane & 15);   // output col for B

    v8f acc = {};
#pragma unroll 4
    for (int ks = 0; ks < 32; ++ks) {
        v16h af = load_frag(hbuf  + (size_t)rA * PROJ_N + ks * 32 + h0);
        v16h bf = load_frag(woutT + (size_t)cB * PROJ_N + ks * 32 + h0);
        acc = __builtin_amdgcn_wmma_f32_16x16x32_f16(
                false, af, false, bf, (short)0, acc, false, false);
    }
    const float bias = obias[cB];
#pragma unroll
    for (int j = 0; j < 8; ++j)
        out[(size_t)(m0 + h0 + j) * 512 + cB] = acc[j] + bias;
}

// ---------------------------------------------------------------------------
extern "C" void kernel_launch(void* const* d_in, const int* in_sizes, int n_in,
                              void* d_out, int out_size, void* d_ws, size_t ws_size,
                              hipStream_t stream) {
    const float* x    = (const float*)d_in[0];
    const float* piw  = (const float*)d_in[1];
    const float* pib  = (const float*)d_in[2];
    const float* w0   = (const float*)d_in[3];
    const float* b0   = (const float*)d_in[4];
    const float* g0w  = (const float*)d_in[5];
    const float* g0b  = (const float*)d_in[6];
    const float* w1   = (const float*)d_in[7];
    const float* c1b  = (const float*)d_in[8];
    const float* g1w  = (const float*)d_in[9];
    const float* g1b  = (const float*)d_in[10];
    const float* w2   = (const float*)d_in[11];
    const float* b2   = (const float*)d_in[12];
    const float* wout = (const float*)d_in[13];
    const float* ob   = (const float*)d_in[14];
    float* out = (float*)d_out;

    unsigned char* ws = (unsigned char*)d_ws;
    float*    partials = (float*)(ws + OFF_PART);
    _Float16* w1f16    = (_Float16*)(ws + OFF_W1);
    _Float16* woutT    = (_Float16*)(ws + OFF_WOUT);
    _Float16* a0t      = (_Float16*)(ws + OFF_A0T);
    float*    c1raw    = (float*)(ws + OFF_C1);
    _Float16* a1t      = (_Float16*)(ws + OFF_A1T);
    _Float16* hbuf     = (_Float16*)(ws + OFF_H);

    // weight prep (tiny, L2-resident thereafter)
    prep_w1_kernel  <<<48,   256, 0, stream>>>(w1, w1f16);
    prep_wout_kernel<<<2048, 256, 0, stream>>>(wout, woutT);

    // HBM-bound streaming reduction of x (the 22.5 us roofline floor)
    reduce_x_kernel <<<B_SZ * 32, 256, 0, stream>>>(x, partials);

    // proj_in + conv0 + gn0 + mish
    stage0_kernel   <<<B_SZ, 256, 0, stream>>>(partials, piw, pib, w0, b0, g0w, g0b, a0t);

    // conv1 via f16 WMMA + gn1 + mish
    conv1_wmma_kernel<<<B_SZ, 256, 0, stream>>>(a0t, w1f16, c1b, g1w, g1b, c1raw, a1t);

    // conv2 -> h (f16)
    conv2_kernel    <<<B_SZ * 4, 256, 0, stream>>>(a1t, w2, b2, hbuf);

    // final GEMM via f16 WMMA + bias
    out_gemm_kernel <<<32, 256, 0, stream>>>(hbuf, woutT, ob, out);
}
